// MultiHeadMaskedSelfAttention_80908593922306
// MI455X (gfx1250) — matrix operand order repaired and checked by reading, not compiled, not run on hardware
//
#include <hip/hip_runtime.h>

// ---------------------------------------------------------------------------
// MultiHeadMaskedSelfAttention for MI455X (gfx1250, wave32, WMMA bf16, TDM)
//   B=4, S=2048, E=1024, H=16, D=64
// Kernel 0: convert X, W fp32 -> bf16 (native cvt)
// Kernel 1: qkv = x @ W^T + b   (double-buffered TDM tile loads into LDS,
//           bf16 WMMA, f32 accumulate — DMA overlapped with WMMA)
// Kernel 2: causal flash attention (bf16 WMMA, fp32 online softmax, exp2)
// ---------------------------------------------------------------------------

#define BATCH 4
#define SEQ   2048
#define EMB   1024
#define NH    16
#define HD    64
#define M_ROWS (BATCH * SEQ)       // 8192
#define N_COLS (3 * EMB)           // 3072
#define K_DIM  EMB                 // 1024

typedef __attribute__((ext_vector_type(16))) __bf16 v16bf;
typedef __attribute__((ext_vector_type(8)))  __bf16 v8bf;
typedef __attribute__((ext_vector_type(4)))  __bf16 v4bf;
typedef __attribute__((ext_vector_type(8)))  float  v8f;
typedef __attribute__((ext_vector_type(4)))  unsigned int v4u;
typedef __attribute__((ext_vector_type(8)))  int v8i;
typedef __attribute__((ext_vector_type(4)))  int v4i;

static __device__ __forceinline__ v8f vzero8() {
  v8f z;
#pragma unroll
  for (int i = 0; i < 8; ++i) z[i] = 0.0f;
  return z;
}

static __device__ __forceinline__ v16bf pack16(v8bf lo, v8bf hi) {
  v16bf r;
#pragma unroll
  for (int i = 0; i < 8; ++i) { r[i] = lo[i]; r[i + 8] = hi[i]; }
  return r;
}

static __device__ __forceinline__ v8f wmma_bf16(v16bf a, v16bf b, v8f c) {
  return __builtin_amdgcn_wmma_f32_16x16x32_bf16(false, a, false, b,
                                                 (short)0, c, false, false);
}

// ---------------------------------------------------------------------------
// Tensor Data Mover: DMA a [rows x 32] bf16 tile (row stride = srcRowLen
// elements) from global memory into LDS at lds_off. 2D D# per ISA ch.8.
// ---------------------------------------------------------------------------
static __device__ __forceinline__ void tdm_load_tile_bf16(
    unsigned lds_off, const __bf16* gptr, unsigned rows, unsigned srcRowLen) {
  unsigned long long ga = (unsigned long long)(uintptr_t)gptr;
  v4u g0;
  g0[0] = 1u;                                     // count=1, user mode
  g0[1] = lds_off;                                // lds_addr (bytes)
  g0[2] = (unsigned)(ga & 0xFFFFFFFFu);           // global_addr[31:0]
  g0[3] = (unsigned)((ga >> 32) & 0x01FFFFFFu)    // global_addr[56:32]
        | (2u << 30);                             // type = 2 ("image")
  v8i g1;
  g1[0] = (int)(1u << 16);                        // data_size=1 -> 2 bytes
  g1[1] = (int)((srcRowLen & 0xFFFFu) << 16);     // tensor_dim0[15:0]
  g1[2] = (int)((srcRowLen >> 16) |               // tensor_dim0[31:16]
                ((unsigned)M_ROWS << 16));        // tensor_dim1[15:0]
  g1[3] = (int)(32u << 16);                       // tile_dim0 = 32
  g1[4] = (int)(rows & 0xFFFFu);                  // tile_dim1 = rows, dim2=0
  g1[5] = (int)srcRowLen;                         // tensor_dim0_stride[31:0]
  g1[6] = 0;                                      // stride hi / dim1_stride lo
  g1[7] = 0;
  v4i z4 = {0, 0, 0, 0};
#if __clang_major__ >= 23
  v8i z8 = {0, 0, 0, 0, 0, 0, 0, 0};
  __builtin_amdgcn_tensor_load_to_lds(g0, g1, z4, z4, z8, 0);
#else
  __builtin_amdgcn_tensor_load_to_lds(g0, g1, z4, z4, 0);
#endif
}

// ---------------------------------------------------------------------------
// Kernel 0: streaming fp32 -> bf16 convert (native v_cvt path).
// ---------------------------------------------------------------------------
__global__ __launch_bounds__(256)
void cvt_bf16_kernel(const float* __restrict__ src, __bf16* __restrict__ dst,
                     int n4) {
  int i = blockIdx.x * 256 + threadIdx.x;
  if (i >= n4) return;
  float4 v = ((const float4*)src)[i];
  v4bf b;
  b[0] = (__bf16)v.x; b[1] = (__bf16)v.y;
  b[2] = (__bf16)v.z; b[3] = (__bf16)v.w;
  ((v4bf*)dst)[i] = b;
}

// ---------------------------------------------------------------------------
// Kernel 1: QKV projection GEMM, double-buffered TDM-fed.
//   Xb [8192 x 1024] bf16, Wb [3072 x 1024] bf16 (W^T column n == W row n)
//   Output (bf16): Kb[bh][s][d], Qb[bh][s][d], Vt[bh][d][s]
// Block tile 128x128, K-step 32, 256 threads = 8 waves, wave tile 64x32.
// TDM for k-slab i+1 is in flight while WMMAs consume slab i.
// ---------------------------------------------------------------------------
__global__ __launch_bounds__(256)
void qkv_gemm_kernel(const __bf16* __restrict__ Xb,
                     const __bf16* __restrict__ Wb,
                     const float* __restrict__ bias,
                     __bf16* __restrict__ Kb, __bf16* __restrict__ Qb,
                     __bf16* __restrict__ Vt) {
  __shared__ __bf16 As[2][128][32];   // [buf][m][k]  8 KB each
  __shared__ __bf16 Bs[2][128][32];   // [buf][n][k]

  const int t    = threadIdx.x;
  const int lane = t & 31;
  const int wave = t >> 5;           // 0..7
  const int wrow = wave & 1;         // m offset 0/64
  const int wcol = wave >> 1;        // n offset 0/32/64/96
  const int ln   = lane & 15;
  const int hi   = lane >> 4;
  const int m0   = blockIdx.x * 128;
  const int n0   = blockIdx.y * 128;

  const unsigned ldsA = (unsigned)(uintptr_t)&As[0][0][0];
  const unsigned ldsB = (unsigned)(uintptr_t)&Bs[0][0][0];
  const unsigned BUFB = 128u * 32u * 2u;   // 8192 bytes per buffer

  const __bf16* Xrow = Xb + (size_t)m0 * K_DIM;
  const __bf16* Wrow = Wb + (size_t)n0 * K_DIM;

  v8f acc[4][2];
#pragma unroll
  for (int mi = 0; mi < 4; ++mi)
#pragma unroll
    for (int ni = 0; ni < 2; ++ni) acc[mi][ni] = vzero8();

  // ---- prologue: DMA k-slab 0 into buffer 0 ----
  if (wave == 0) {
    tdm_load_tile_bf16(ldsA, Xrow, 128, K_DIM);
    tdm_load_tile_bf16(ldsB, Wrow, 128, K_DIM);
  }

  const int NITER = K_DIM / 32;            // 32
  for (int it = 0; it < NITER; ++it) {
    const int cur = it & 1;
    if (wave == 0) {
      if (it + 1 < NITER) {
        // issue next slab into the other buffer (consumed 2 barriers ago),
        // then wait only for the *current* slab (in-order TDM completion).
        const int kc = (it + 1) * 32;
        const unsigned off = (unsigned)(cur ^ 1) * BUFB;
        tdm_load_tile_bf16(ldsA + off, Xrow + kc, 128, K_DIM);
        tdm_load_tile_bf16(ldsB + off, Wrow + kc, 128, K_DIM);
        __builtin_amdgcn_s_wait_tensorcnt(2);
      } else {
        __builtin_amdgcn_s_wait_tensorcnt(0);
      }
    }
    __syncthreads();

    // ---- A fragments: 16x32 bf16, lane = row, K split per lane-half ----
    v16bf af[4];
    const int g = hi * 16;
#pragma unroll
    for (int mi = 0; mi < 4; ++mi) {
      int m = wrow * 64 + mi * 16 + ln;
      v8bf lo = *(const v8bf*)&As[cur][m][g];        // K = g..g+7
      v8bf hh = *(const v8bf*)&As[cur][m][g + 8];    // K = g+8..g+15
      af[mi] = pack16(lo, hh);
    }
    // ---- B fragments: 32x16 bf16, lane = column, contiguous 16 K-run ----
    v16bf bf_[2];
#pragma unroll
    for (int ni = 0; ni < 2; ++ni) {
      int n = wcol * 32 + ni * 16 + ln;
      bf_[ni] = *(const v16bf*)&Bs[cur][n][hi * 16]; // K = hi*16..hi*16+15
    }
#pragma unroll
    for (int mi = 0; mi < 4; ++mi)
#pragma unroll
      for (int ni = 0; ni < 2; ++ni)
        acc[mi][ni] = wmma_bf16(af[mi], bf_[ni], acc[mi][ni]);

    __syncthreads();
  }

  // ---- epilogue: + bias, scatter bf16 into K / Q / Vt layouts ----
#pragma unroll
  for (int ni = 0; ni < 2; ++ni) {
    int gcol = n0 + wcol * 32 + ni * 16 + ln;   // 0..3071
    float bv = bias[gcol];
    int csel = gcol >> 10;                      // 0=K, 1=Q, 2=V
    int h    = (gcol >> 6) & 15;
    int d    = gcol & 63;
#pragma unroll
    for (int mi = 0; mi < 4; ++mi) {
#pragma unroll
      for (int r = 0; r < 8; ++r) {
        int grow = m0 + wrow * 64 + mi * 16 + r + (hi << 3);
        int b = grow >> 11;
        int s = grow & 2047;
        size_t bh = (size_t)(b * NH + h);
        __bf16 val = (__bf16)(acc[mi][ni][r] + bv);
        if (csel == 0)      Kb[(bh * SEQ + s) * HD + d] = val;
        else if (csel == 1) Qb[(bh * SEQ + s) * HD + d] = val;
        else                Vt[(bh * HD + d) * SEQ + s] = val;
      }
    }
  }
}

// ---------------------------------------------------------------------------
// Kernel 2: causal flash attention.
//   grid = (32 query-blocks of 64, 64 bh), block = 128 threads = 4 waves.
//   Each wave owns 16 query rows; keys walked in blocks of 32 (causal bound).
//   Softmax in base-2: scale = (1/sqrt(64)) * log2(e) folded into scores.
// ---------------------------------------------------------------------------
#define SCALE_LOG2E 0.1803368801111204f   // 0.125 * 1.4426950408889634

__global__ __launch_bounds__(128)
void flash_attn_kernel(const __bf16* __restrict__ Kb,
                       const __bf16* __restrict__ Qb,
                       const __bf16* __restrict__ Vt,
                       float* __restrict__ out) {
  __shared__ __bf16 Plds[4][16][32];           // per-wave P staging

  const int lane  = threadIdx.x & 31;
  const int wave  = threadIdx.x >> 5;          // 0..3
  const int ln    = lane & 15;
  const int hi    = lane >> 4;
  const int bh    = blockIdx.y;                // 0..63
  const int qbase = blockIdx.x * 64 + wave * 16;
  const int g     = hi * 16;

  // Q fragments (A layout), 2 chunks over D=64
  v16bf qf[2];
  {
    const __bf16* qp = Qb + ((size_t)bh * SEQ + (qbase + ln)) * HD;
#pragma unroll
    for (int c = 0; c < 2; ++c) {
      v8bf lo = *(const v8bf*)(qp + c * 32 + g);
      v8bf hh = *(const v8bf*)(qp + c * 32 + g + 8);
      qf[c] = pack16(lo, hh);
    }
  }

  float mrow[8], lrow[8];
  v8f o[4];
#pragma unroll
  for (int r = 0; r < 8; ++r) { mrow[r] = -__builtin_inff(); lrow[r] = 0.0f; }
#pragma unroll
  for (int dt = 0; dt < 4; ++dt) o[dt] = vzero8();

  const int nj = (qbase + 47) >> 5;            // causal: key blocks needed
  for (int j = 0; j < nj; ++j) {
    const int kb = j * 32;

    // ---- scores S = Q K^T : 2 key-tiles x 2 d-chunks ----
    v8f sc[2] = {vzero8(), vzero8()};
#pragma unroll
    for (int tt = 0; tt < 2; ++tt) {
      const __bf16* kp = Kb + ((size_t)bh * SEQ + (kb + tt * 16 + ln)) * HD;
#pragma unroll
      for (int c = 0; c < 2; ++c) {
        v16bf kf = *(const v16bf*)(kp + c * 32 + hi * 16);
        sc[tt] = wmma_bf16(qf[c], kf, sc[tt]);
      }
    }
    // prefetch next key block while softmax runs
    if (kb + 32 < qbase + 16)
      __builtin_prefetch(Kb + ((size_t)bh * SEQ + (kb + 32 + lane)) * HD, 0, 3);

    // ---- scale (into log2 domain) + causal mask + online softmax ----
    float rmax[8];
#pragma unroll
    for (int r = 0; r < 8; ++r) {
      int qi = qbase + r + (hi << 3);
      float best = -__builtin_inff();
#pragma unroll
      for (int tt = 0; tt < 2; ++tt) {
        int kj = kb + tt * 16 + ln;
        float sv = sc[tt][r] * SCALE_LOG2E;
        if (kj > qi) sv = -__builtin_inff();
        sc[tt][r] = sv;
        best = fmaxf(best, sv);
      }
      rmax[r] = best;
    }
#pragma unroll
    for (int msk = 1; msk < 16; msk <<= 1)
#pragma unroll
      for (int r = 0; r < 8; ++r)
        rmax[r] = fmaxf(rmax[r], __shfl_xor(rmax[r], msk, 32));

    float alpha[8], rsum[8];
#pragma unroll
    for (int r = 0; r < 8; ++r) {
      float mn = fmaxf(mrow[r], rmax[r]);
      alpha[r] = __builtin_exp2f(mrow[r] - mn);   // V_EXP_F32 is base-2 native
      mrow[r]  = mn;
    }
#pragma unroll
    for (int r = 0; r < 8; ++r) {
      float p0 = __builtin_exp2f(sc[0][r] - mrow[r]);
      float p1 = __builtin_exp2f(sc[1][r] - mrow[r]);
      sc[0][r] = p0; sc[1][r] = p1;
      rsum[r] = p0 + p1;
    }
#pragma unroll
    for (int msk = 1; msk < 16; msk <<= 1)
#pragma unroll
      for (int r = 0; r < 8; ++r)
        rsum[r] += __shfl_xor(rsum[r], msk, 32);
#pragma unroll
    for (int r = 0; r < 8; ++r) lrow[r] = lrow[r] * alpha[r] + rsum[r];
#pragma unroll
    for (int dt = 0; dt < 4; ++dt)
#pragma unroll
      for (int r = 0; r < 8; ++r) o[dt][r] *= alpha[r];

    // ---- P: C-layout -> LDS -> A-layout (per-wave region, in-order DS) ----
#pragma unroll
    for (int tt = 0; tt < 2; ++tt)
#pragma unroll
      for (int r = 0; r < 8; ++r)
        Plds[wave][r + (hi << 3)][tt * 16 + ln] = (__bf16)sc[tt][r];
    asm volatile("s_wait_dscnt 0" ::: "memory");
    v16bf pf;
    {
      v8bf lo = *(const v8bf*)&Plds[wave][ln][g];
      v8bf hh = *(const v8bf*)&Plds[wave][ln][g + 8];
      pf = pack16(lo, hh);
    }

    // ---- O += P @ V : V fragments contiguous thanks to Vt[bh][d][s] ----
#pragma unroll
    for (int dt = 0; dt < 4; ++dt) {
      const __bf16* vp =
          Vt + ((size_t)bh * HD + (dt * 16 + ln)) * SEQ + kb + hi * 16;
      v16bf vf = *(const v16bf*)vp;
      o[dt] = wmma_bf16(pf, vf, o[dt]);
    }
  }

  // ---- epilogue: normalize, write fp32 out[b][s][h*64+d] ----
  const int b = bh >> 4;
  const int h = bh & 15;
#pragma unroll
  for (int r = 0; r < 8; ++r) {
    int qi = qbase + r + (hi << 3);
    float inv = 1.0f / lrow[r];
    size_t base = ((size_t)b * SEQ + qi) * EMB + h * HD;
#pragma unroll
    for (int dt = 0; dt < 4; ++dt)
      out[base + dt * 16 + ln] = o[dt][r] * inv;
  }
}

// ---------------------------------------------------------------------------
extern "C" void kernel_launch(void* const* d_in, const int* in_sizes, int n_in,
                              void* d_out, int out_size, void* d_ws,
                              size_t ws_size, hipStream_t stream) {
  (void)in_sizes; (void)n_in; (void)out_size; (void)ws_size;
  const float* X    = (const float*)d_in[0];
  const float* W    = (const float*)d_in[1];
  const float* bias = (const float*)d_in[2];
  float* out        = (float*)d_out;

  const size_t XB_BYTES  = (size_t)M_ROWS * K_DIM * 2;  // 16 MB
  const size_t WB_BYTES  = (size_t)N_COLS * K_DIM * 2;  //  6 MB
  const size_t QKV_BYTES = (size_t)M_ROWS * EMB * 2;    // 16 MB each
  char* ws = (char*)d_ws;
  __bf16* Xb = (__bf16*)(ws);
  __bf16* Wb = (__bf16*)(ws + XB_BYTES);
  __bf16* Kb = (__bf16*)(ws + XB_BYTES + WB_BYTES);
  __bf16* Qb = (__bf16*)(ws + XB_BYTES + WB_BYTES + QKV_BYTES);
  __bf16* Vt = (__bf16*)(ws + XB_BYTES + WB_BYTES + 2 * QKV_BYTES);

  // Kernel 0: convert inputs to bf16
  {
    int n4x = (M_ROWS * K_DIM) / 4;                     // 2,097,152
    int n4w = (N_COLS * K_DIM) / 4;                     //   786,432
    cvt_bf16_kernel<<<dim3((n4x + 255) / 256), dim3(256), 0, stream>>>(X, Xb, n4x);
    cvt_bf16_kernel<<<dim3((n4w + 255) / 256), dim3(256), 0, stream>>>(W, Wb, n4w);
  }

  dim3 g1(M_ROWS / 128, N_COLS / 128), b1(256);
  qkv_gemm_kernel<<<g1, b1, 0, stream>>>(Xb, Wb, bias, Kb, Qb, Vt);

  dim3 g2(SEQ / 64, BATCH * NH), b2(128);
  flash_attn_kernel<<<g2, b2, 0, stream>>>(Kb, Qb, Vt, out);
}
